// SpikingJellyGCN_25202868093204
// MI455X (gfx1250) — matrix operand order repaired and checked by reading, not compiled, not run on hardware
//
#include <hip/hip_runtime.h>
#include <hip/hip_bf16.h>
#include <stdint.h>
#include <stddef.h>

typedef __attribute__((ext_vector_type(16))) _Float16 v16h;
typedef __attribute__((ext_vector_type(8)))  _Float16 v8h;
typedef __attribute__((ext_vector_type(8)))  float    v8f;

#define N_SAMP   16384
#define IN_DIM   1024
#define OUT_DIM  256
#define T_STEPS  32
#define KEEP_P   0.4f

// ---------------------------------------------------------------------------
// Exact JAX threefry2x32 block cipher (key schedule per jax._src.prng).
// ---------------------------------------------------------------------------
__host__ __device__ inline void tf2x32(uint32_t k0, uint32_t k1,
                                       uint32_t c0, uint32_t c1,
                                       uint32_t &o0, uint32_t &o1) {
  const uint32_t ks2 = k0 ^ k1 ^ 0x1BD11BDAu;
  uint32_t x0 = c0 + k0, x1 = c1 + k1;
#define TF_R(r) { x0 += x1; x1 = (x1 << (r)) | (x1 >> (32 - (r))); x1 ^= x0; }
  TF_R(13) TF_R(15) TF_R(26) TF_R(6)   x0 += k1;  x1 += ks2 + 1u;
  TF_R(17) TF_R(29) TF_R(16) TF_R(24)  x0 += ks2; x1 += k0 + 2u;
  TF_R(13) TF_R(15) TF_R(26) TF_R(6)   x0 += k0;  x1 += k1 + 3u;
  TF_R(17) TF_R(29) TF_R(16) TF_R(24)  x0 += k1;  x1 += ks2 + 4u;
  TF_R(13) TF_R(15) TF_R(26) TF_R(6)   x0 += ks2; x1 += k0 + 5u;
#undef TF_R
  o0 = x0; o1 = x1;
}

__device__ inline float bits_to_unit(uint32_t b) {
  // JAX uniform: bitcast((bits>>9)|0x3f800000, f32) - 1.0  ->  [0,1)
  return __uint_as_float((b >> 9) | 0x3f800000u) - 1.0f;
}

struct DropKeys { uint32_t k0[T_STEPS]; uint32_t k1[T_STEPS]; };

// ---------------------------------------------------------------------------
// Kernel 1: Poisson encode at t=0 (spikes are constant for t>=1 because the
// scan carry re-feeds binary spikes through u<x). Output f16 {0,1}.
// JAX pairs flat counts (j, j+half) through one cipher block.
// ---------------------------------------------------------------------------
__global__ void __launch_bounds__(256)
snn_encode(const float* __restrict__ x, _Float16* __restrict__ spk,
           uint32_t k0, uint32_t k1) {
  const uint32_t HALF = (uint32_t)(N_SAMP * IN_DIM) / 2u;   // 8388608
  uint32_t j = blockIdx.x * 256u + threadIdx.x;
  if (j >= HALF) return;
  uint32_t b0, b1;
  tf2x32(k0, k1, j, j + HALF, b0, b1);
  spk[j]        = (bits_to_unit(b0) < x[j])        ? (_Float16)1.0f : (_Float16)0.0f;
  spk[j + HALF] = (bits_to_unit(b1) < x[j + HALF]) ? (_Float16)1.0f : (_Float16)0.0f;
}

// Kernel 1b: W (f32, row-major OUT x IN) -> f16.
__global__ void __launch_bounds__(256)
snn_wconv(const float* __restrict__ W, _Float16* __restrict__ wh) {
  int i = blockIdx.x * 256 + threadIdx.x;
  if (i < OUT_DIM * IN_DIM) wh[i] = (_Float16)W[i];
}

// ---------------------------------------------------------------------------
// Kernel 2: h_pre = spk @ W^T + b via v_wmma_f32_16x16x32_f16.
// One wave -> 16 rows x 64 cols (4 accumulator tiles, A reused 4x).
// Software-pipelined with unroll-2 steady state so the double buffers
// ping-pong by copy propagation (no v_mov rotation), and each WMMA group
// waits only on the *previous* iteration's load clause (partial loadcnt).
// ---------------------------------------------------------------------------
__device__ __forceinline__ v16h lda(const _Float16* __restrict__ arow,
                                    int kb, int hi) {
  v8h alo = *(const v8h*)(arow + kb + hi * 16);
  v8h ahi = *(const v8h*)(arow + kb + hi * 16 + 8);
  v16h a;
#pragma unroll
  for (int i = 0; i < 8; ++i) { a[i] = alo[i]; a[8 + i] = ahi[i]; }
  return a;
}

__global__ void __launch_bounds__(256)
snn_gemm(const _Float16* __restrict__ spk, const _Float16* __restrict__ wh,
         const float* __restrict__ bias, float* __restrict__ hpre) {
  const int wave = (int)((blockIdx.x * 256u + threadIdx.x) >> 5);
  const int lane = (int)(threadIdx.x & 31u);
  const int rt   = wave >> 2;        // row tile 0..1023 (16 rows each)
  const int cg   = wave & 3;         // col group 0..3   (64 cols each)
  const int m    = lane & 15;
  const int hi   = lane >> 4;        // lane half selector
  const int nloc = lane & 15;

  const _Float16* __restrict__ arow = spk + (size_t)(rt * 16 + m) * IN_DIM;
  const _Float16* __restrict__ bcol = wh  + (size_t)(cg * 64 + nloc) * IN_DIM;

  v8f c0 = {}, c1 = {}, c2 = {}, c3 = {};

  // Prologue: load k-step 0.
  v16h a  = lda(arow, 0, hi);
  const _Float16* bb = bcol + hi * 16;
  v16h b0 = *(const v16h*)(bb);
  v16h b1 = *(const v16h*)(bb + 16 * IN_DIM);
  v16h b2 = *(const v16h*)(bb + 32 * IN_DIM);
  v16h b3 = *(const v16h*)(bb + 48 * IN_DIM);

#pragma unroll 2
  for (int kb = 32; kb < IN_DIM; kb += 32) {
    v16h an = lda(arow, kb, hi);
    const _Float16* bn = bcol + kb + hi * 16;
    v16h n0 = *(const v16h*)(bn);
    v16h n1 = *(const v16h*)(bn + 16 * IN_DIM);
    v16h n2 = *(const v16h*)(bn + 32 * IN_DIM);
    v16h n3 = *(const v16h*)(bn + 48 * IN_DIM);

    c0 = __builtin_amdgcn_wmma_f32_16x16x32_f16(false, a, false, b0, (short)0, c0, false, false);
    c1 = __builtin_amdgcn_wmma_f32_16x16x32_f16(false, a, false, b1, (short)0, c1, false, false);
    c2 = __builtin_amdgcn_wmma_f32_16x16x32_f16(false, a, false, b2, (short)0, c2, false, false);
    c3 = __builtin_amdgcn_wmma_f32_16x16x32_f16(false, a, false, b3, (short)0, c3, false, false);

    a = an; b0 = n0; b1 = n1; b2 = n2; b3 = n3;
  }
  c0 = __builtin_amdgcn_wmma_f32_16x16x32_f16(false, a, false, b0, (short)0, c0, false, false);
  c1 = __builtin_amdgcn_wmma_f32_16x16x32_f16(false, a, false, b1, (short)0, c1, false, false);
  c2 = __builtin_amdgcn_wmma_f32_16x16x32_f16(false, a, false, b2, (short)0, c2, false, false);
  c3 = __builtin_amdgcn_wmma_f32_16x16x32_f16(false, a, false, b3, (short)0, c3, false, false);

  const float bb0 = bias[cg * 64 +  0 + nloc];
  const float bb1 = bias[cg * 64 + 16 + nloc];
  const float bb2 = bias[cg * 64 + 32 + nloc];
  const float bb3 = bias[cg * 64 + 48 + nloc];
#pragma unroll
  for (int r = 0; r < 8; ++r) {
    const int row = rt * 16 + r + 8 * hi;   // D: VGPR r -> M=r(+8 for hi lanes)
    float* __restrict__ orow = hpre + (size_t)row * OUT_DIM + cg * 64 + nloc;
    orow[ 0] = c0[r] + bb0;
    orow[16] = c1[r] + bb1;
    orow[32] = c2[r] + bb2;
    orow[48] = c3[r] + bb3;
  }
}

// ---------------------------------------------------------------------------
// Kernel 3: 32-step LIF + dropout + row softmax, processing row pair
// (n, n+8192) per block. Elements j and j+HALF share one threefry block in
// JAX's halves-split counter scheme (HALF = 8192*256), so one cipher per
// timestep drives BOTH neurons -> half the hash work of the naive mapping.
// ---------------------------------------------------------------------------
__global__ void __launch_bounds__(256)
snn_lif_softmax(float* __restrict__ io, DropKeys dk) {
  const int n  = blockIdx.x;               // 0..8191
  const int n2 = n + (N_SAMP / 2);         // paired row
  const int o  = (int)threadIdx.x;
  const uint32_t j    = (uint32_t)n * OUT_DIM + (uint32_t)o;   // always < HALF
  const uint32_t HALF = (uint32_t)(N_SAMP * OUT_DIM) / 2u;     // 2097152

  const float ha = io[(size_t)n  * OUT_DIM + o];
  const float hb = io[(size_t)n2 * OUT_DIM + o];
  float va = 0.0f, aa = 0.0f;
  float vb = 0.0f, ab = 0.0f;
#pragma unroll 4
  for (int t = 0; t < T_STEPS; ++t) {
    uint32_t r0, r1;
    tf2x32(dk.k0[t], dk.k1[t], j, j + HALF, r0, r1);

    const float ua = bits_to_unit(r0);
    const float hva = (ua < KEEP_P) ? ha * (1.0f / KEEP_P) : 0.0f;
    va = va + (hva - va) / 10.0f;
    const float sa = (va - 1.0f >= 0.0f) ? 1.0f : 0.0f;
    aa += sa;
    va = (1.0f - sa) * va;

    const float ub = bits_to_unit(r1);
    const float hvb = (ub < KEEP_P) ? hb * (1.0f / KEEP_P) : 0.0f;
    vb = vb + (hvb - vb) / 10.0f;
    const float sb = (vb - 1.0f >= 0.0f) ? 1.0f : 0.0f;
    ab += sb;
    vb = (1.0f - sb) * vb;
  }

  __shared__ float red[2][OUT_DIM];
  red[0][o] = aa; red[1][o] = ab; __syncthreads();
#pragma unroll
  for (int sft = OUT_DIM / 2; sft > 0; sft >>= 1) {
    if (o < sft) {
      red[0][o] = fmaxf(red[0][o], red[0][o + sft]);
      red[1][o] = fmaxf(red[1][o], red[1][o + sft]);
    }
    __syncthreads();
  }
  const float mxa = red[0][0], mxb = red[1][0]; __syncthreads();
  const float ea = expf(aa - mxa), eb = expf(ab - mxb);
  red[0][o] = ea; red[1][o] = eb; __syncthreads();
#pragma unroll
  for (int sft = OUT_DIM / 2; sft > 0; sft >>= 1) {
    if (o < sft) {
      red[0][o] += red[0][o + sft];
      red[1][o] += red[1][o + sft];
    }
    __syncthreads();
  }
  io[(size_t)n  * OUT_DIM + o] = ea / red[0][0];
  io[(size_t)n2 * OUT_DIM + o] = eb / red[1][0];
}

// ---------------------------------------------------------------------------
extern "C" void kernel_launch(void* const* d_in, const int* in_sizes, int n_in,
                              void* d_out, int out_size, void* d_ws, size_t ws_size,
                              hipStream_t stream) {
  (void)in_sizes; (void)n_in; (void)out_size; (void)ws_size;
  const float* x = (const float*)d_in[0];     // (16384, 1024) f32
  const float* W = (const float*)d_in[1];     // (256, 1024)  f32
  const float* b = (const float*)d_in[2];     // (256,)       f32
  float* out = (float*)d_out;                 // (16384, 256) f32

  _Float16* spk = (_Float16*)d_ws;                                         // 32 MB
  _Float16* wh  = (_Float16*)((char*)d_ws +
                              (size_t)N_SAMP * IN_DIM * sizeof(_Float16)); // +0.5 MB

  // Host-side key derivation (depends only on literals -> deterministic):
  // base = key(1234) = (0,1234); per t: k = fold_in(base,t); k_enc,k_drop = split(k)
  // split: cipher pairs (0,2),(1,3); k_enc=(o0a,o0b), k_drop=(o1a,o1b)
  uint32_t kenc0 = 0, kenc1 = 0;
  DropKeys dk;
  for (int t = 0; t < T_STEPS; ++t) {
    uint32_t f0, f1, a0, a1, c0, c1;
    tf2x32(0u, 1234u, 0u, (uint32_t)t, f0, f1);   // fold_in
    tf2x32(f0, f1, 0u, 2u, a0, a1);
    tf2x32(f0, f1, 1u, 3u, c0, c1);
    if (t == 0) { kenc0 = a0; kenc1 = c0; }
    dk.k0[t] = a1; dk.k1[t] = c1;
  }

  snn_encode<<<(N_SAMP * IN_DIM / 2) / 256, 256, 0, stream>>>(x, spk, kenc0, kenc1);
  snn_wconv<<<(OUT_DIM * IN_DIM) / 256, 256, 0, stream>>>(W, wh);
  // 1024 row tiles * 4 col groups = 4096 waves; 8 waves/block -> 512 blocks
  snn_gemm<<<512, 256, 0, stream>>>(spk, wh, b, out);
  // Row-paired LIF: 8192 blocks cover rows (n, n+8192)
  snn_lif_softmax<<<N_SAMP / 2, 256, 0, stream>>>(out, dk);
}
